// Dynamic_conv2d_24446953849520
// MI455X (gfx1250) — matrix operand order repaired and checked by reading, not compiled, not run on hardware
//
#include <hip/hip_runtime.h>

// ---------------------------------------------------------------------------
// Dynamic conv2d (CondConv-style) for MI455X / gfx1250.
// wave32 + v_wmma_f32_16x16x32_f16 + Tensor Data Mover (tensor_load_to_lds)
// double-buffered LDS, 2 output rows per WG (A-frag reused for 4 WMMAs,
// halves weight-tile L2 traffic to ~0.6 GB).
//
// Heavy op = per-sample implicit GEMM: M=256(oc), N=4096(pix), K=2304(ic*9).
// 77.3 GFLOP; HBM floor ~145MB (~6us @23.3TB/s) => matrix-compute bound.
// ---------------------------------------------------------------------------

typedef __attribute__((ext_vector_type(16))) _Float16 v16h;
typedef __attribute__((ext_vector_type(8)))  float    v8f;
typedef unsigned int v4u __attribute__((ext_vector_type(4)));
typedef int          v4i __attribute__((ext_vector_type(4)));
typedef int          v8i __attribute__((ext_vector_type(8)));

#define NB         16
#define IN_PLANES  256
#define OUT_PLANES 256
#define KBANK      4
#define HIDDEN     65
#define HW         64
#define NPIX       4096
#define TEMP       34.0f
#define XPITCH     40   // halves per LDS row: 64B data + 16B TDM pad (16B-aligned)

union Frag16 { uint4 q[2]; v16h v; };

__device__ __forceinline__ v16h ldsLoadFrag(const _Float16* p) {
  Frag16 f;
  f.q[0] = *(const uint4*)(p);
  f.q[1] = *(const uint4*)(p + 8);
  return f.v;
}

// ---------------------------------------------------------------------------
// TDM: build D# per ISA 8.3/8.4 and issue tensor_load_to_lds.
// data_size=2B. Pad: after each 16 data-DWORDs (one 64B row) skip 4 DWORDs in
// LDS (pad_interval=3, pad_amount=3) -> 80B row pitch == XPITCH halves.
// ---------------------------------------------------------------------------
__device__ __forceinline__ void tdm_load(
    unsigned long long gaddr, unsigned int ldsaddr,
    unsigned int tdim0, unsigned long long stride0, unsigned int tile0,
    unsigned int tdim1, unsigned long long stride1, unsigned int tile1,
    unsigned int tdim2, unsigned int tile2) {
  v4u g0;
  g0.x = 1u;                                            // count=1 (valid user D#)
  g0.y = ldsaddr;                                       // lds_addr (bytes)
  g0.z = (unsigned int)gaddr;                           // global_addr[31:0]
  g0.w = (unsigned int)((gaddr >> 32) & 0x01ffffffull)  // global_addr[56:32]
       | (2u << 30);                                    // type=2 ("image")

  v8i g1;
  g1[0] = (int)((1u << 16)                              // data_size = 2 bytes
              | (1u << 20)                              // pad_enable
              | (3u << 22)                              // pad_interval: 16 DW
              | (3u << 25));                            // pad_amount: 4 DW
  g1[1] = (int)((tdim0 & 0xffffu) << 16);
  g1[2] = (int)((tdim0 >> 16) | ((tdim1 & 0xffffu) << 16));
  g1[3] = (int)((tdim1 >> 16) | (tile0 << 16));
  g1[4] = (int)((tile1 & 0xffffu) | (tile2 << 16));
  g1[5] = (int)(unsigned int)(stride0 & 0xffffffffull);
  g1[6] = (int)((unsigned int)((stride0 >> 32) & 0xffffull)
              | ((unsigned int)(stride1 & 0xffffull) << 16));
  g1[7] = (int)(unsigned int)((stride1 >> 16) & 0xffffffffull);

  v4i g2; g2[0] = (int)tdim2; g2[1] = 0; g2[2] = 0; g2[3] = 0;
  v4i g3; g3[0] = 0; g3[1] = 0; g3[2] = 0; g3[3] = 0;

#if defined(__clang_major__) && (__clang_major__ >= 23)
  v8i g4; g4[0]=0; g4[1]=0; g4[2]=0; g4[3]=0; g4[4]=0; g4[5]=0; g4[6]=0; g4[7]=0;
  __builtin_amdgcn_tensor_load_to_lds(g0, g1, g2, g3, g4, 0);
#else
  __builtin_amdgcn_tensor_load_to_lds(g0, g1, g2, g3, 0);
#endif
}

// ---------------------------------------------------------------------------
// Kernel 1: attention (one block per sample)
// ---------------------------------------------------------------------------
__global__ __launch_bounds__(256) void attention_kernel(
    const float* __restrict__ x, const float* __restrict__ w1,
    const float* __restrict__ w2, const float* __restrict__ b2,
    float* __restrict__ att) {
  const int b = blockIdx.x, t = threadIdx.x;
  __shared__ float pooled[IN_PLANES];
  __shared__ float hbuf[HIDDEN];

  const float* xc = x + ((size_t)b * IN_PLANES + t) * NPIX;
  float acc = 0.f;
  for (int p = 0; p < NPIX; p += 4) {
    float4 v = *(const float4*)(xc + p);
    acc += v.x + v.y + v.z + v.w;
  }
  pooled[t] = acc * (1.0f / (float)NPIX);
  __syncthreads();

  if (t < HIDDEN) {
    float s = 0.f;
    for (int c = 0; c < IN_PLANES; ++c) s += pooled[c] * w1[t * IN_PLANES + c];
    hbuf[t] = s > 0.f ? s : 0.f;
  }
  __syncthreads();

  if (t == 0) {
    float lg[KBANK], m = -1e30f;
    for (int k = 0; k < KBANK; ++k) {
      float s = b2[k];
      for (int j = 0; j < HIDDEN; ++j) s += hbuf[j] * w2[k * HIDDEN + j];
      lg[k] = s * (1.0f / TEMP);
      m = fmaxf(m, lg[k]);
    }
    float den = 0.f;
    for (int k = 0; k < KBANK; ++k) { lg[k] = __expf(lg[k] - m); den += lg[k]; }
    for (int k = 0; k < KBANK; ++k) att[b * KBANK + k] = lg[k] / den;
  }
}

// ---------------------------------------------------------------------------
// Kernel 2: aggregate kernel bank -> f16, layout [b][kpos][oc][ic]
// ---------------------------------------------------------------------------
__global__ __launch_bounds__(256) void aggregate_kernel(
    const float* __restrict__ weight, const float* __restrict__ att,
    _Float16* __restrict__ wq) {
  const int oc = blockIdx.x, b = blockIdx.y, ic = threadIdx.x;
  float a[KBANK];
#pragma unroll
  for (int k = 0; k < KBANK; ++k) a[k] = att[b * KBANK + k];
  float s[9];
#pragma unroll
  for (int p = 0; p < 9; ++p) s[p] = 0.f;
#pragma unroll
  for (int k = 0; k < KBANK; ++k) {
    const float* wp = weight + (((size_t)k * OUT_PLANES + oc) * IN_PLANES + ic) * 9;
#pragma unroll
    for (int p = 0; p < 9; ++p) s[p] += a[k] * wp[p];
  }
#pragma unroll
  for (int p = 0; p < 9; ++p)
    wq[(((size_t)b * 9 + p) * OUT_PLANES + oc) * IN_PLANES + ic] = (_Float16)s[p];
}

// ---------------------------------------------------------------------------
// Kernel 3: x f32 NCHW -> f16 NHWC via LDS-tiled transpose
// ---------------------------------------------------------------------------
__global__ __launch_bounds__(256) void to_nhwc_f16(
    const float* __restrict__ x, _Float16* __restrict__ xh) {
  const int h = blockIdx.x, cc = blockIdx.y, b = blockIdx.z;
  const int c0 = cc * 64, t = threadIdx.x;
  __shared__ float tile[64][65];
  {
    const int row = t >> 4, seg = t & 15;
    const float* xp = x + (((size_t)b * IN_PLANES + c0) * HW + h) * HW;
#pragma unroll
    for (int rr = 0; rr < 4; ++rr) {
      const int ci = row + rr * 16;
      float4 v = *(const float4*)(xp + (size_t)ci * NPIX + seg * 4);
      tile[ci][seg * 4 + 0] = v.x; tile[ci][seg * 4 + 1] = v.y;
      tile[ci][seg * 4 + 2] = v.z; tile[ci][seg * 4 + 3] = v.w;
    }
  }
  __syncthreads();
  {
    const int w = t >> 2, cs = t & 3;
    union { uint4 q[2]; _Float16 hb[16]; } u;
#pragma unroll
    for (int j = 0; j < 16; ++j) u.hb[j] = (_Float16)tile[cs * 16 + j][w];
    _Float16* dst = xh + (((size_t)b * HW + h) * HW + w) * IN_PLANES + c0 + cs * 16;
    *(uint4*)dst       = u.q[0];
    *(uint4*)(dst + 8) = u.q[1];
  }
}

// ---------------------------------------------------------------------------
// Kernel 4: implicit-GEMM conv. block=(2 output rows, 64-oc tile, b).
// 256 thr = 8 waves; wave = (mt = wid&3, n-pair = wid>>2); 4 accumulators
// (2 rows x 2 n-tiles) sharing one A fragment. Double-buffered LDS; wave 0
// drives the TDM. Chunk ci in 0..23 == (kh = ci/8, ic0 = (ci%8)*32).
// ---------------------------------------------------------------------------
__global__ __launch_bounds__(256) void dynconv_wmma_kernel(
    const _Float16* __restrict__ xh, const _Float16* __restrict__ wq,
    float* __restrict__ out) {
  const int hb = blockIdx.x, oct = blockIdx.y, b = blockIdx.z;
  const int h0 = hb * 2;
  const int t = threadIdx.x, lane = t & 31, wid = t >> 5;
  const int mt = wid & 3;            // oc tile (16 rows)
  const int np = (wid >> 2) * 32;    // pixel base of n-tile pair

  // A: [kw 3][oc 64][XPITCH] per buffer (TDM 3D tile, natural row order)
  __shared__ __align__(128) _Float16 AfB[2][3 * 64 * XPITCH];     // 2 x 15360 B
  // X: per buffer, 2 row-slabs [w' 66][XPITCH]
  __shared__ __align__(128) _Float16 XtB[2][2][66 * XPITCH];      // 4 x  5280 B

  // zero the padding rows (w'=-1 and w'=64) of all 4 Xt slabs once
  if (t < 32) {
    const int bufz = t >> 4, rowz = (t >> 3) & 1, colz = (t >> 2) & 1, seg = t & 3;
    *(uint4*)(&XtB[bufz][rowz][(colz ? 65 * XPITCH : 0) + seg * 8]) =
        make_uint4(0, 0, 0, 0);
  }

  const unsigned long long wq_base = (unsigned long long)(uintptr_t)wq;
  const unsigned long long xh_base = (unsigned long long)(uintptr_t)xh;

  auto issue = [&](int ci, int buf) -> int {
    const int kh = ci >> 3, ic0 = (ci & 7) * 32;
    const int r0 = h0 + kh - 1;          // r0 in [-1,63]: only low bound can fail
    const int r1 = r0 + 1;               // r1 in [0,64]:  only high bound can fail
    const bool rin0 = (r0 >= 0);
    const bool rin1 = (r1 < HW);
    if (wid == 0) {
      // A: 3D tile [ic=32][oc=64][p=3] of wq[b] at (p=kh*3, oc=oct*64, ic=ic0)
      unsigned long long ga = wq_base + 2ull *
          ((((unsigned long long)b * 9 + kh * 3) * OUT_PLANES + (unsigned)(oct * 64)) * IN_PLANES + (unsigned)ic0);
      tdm_load(ga, (unsigned int)(uintptr_t)&AfB[buf][0],
               IN_PLANES, 256ull, 32,        // dim0: ic   (len, line-stride, tile)
               OUT_PLANES, 65536ull, 64,     // dim1: oc   (len, plane-stride, tile)
               3, 3);                        // dim2: p
      // B: 2D tiles [ic=32][w=64] of NHWC xh at (b, r, 0, ic0)
      if (rin0) {
        unsigned long long gx = xh_base + 2ull *
            ((((unsigned long long)b * HW + (unsigned)r0) * HW) * IN_PLANES + (unsigned)ic0);
        tdm_load(gx, (unsigned int)(uintptr_t)&XtB[buf][0][XPITCH],
                 IN_PLANES, 256ull, 32, HW, 0ull, 64, 0, 0);
      }
      if (rin1) {
        unsigned long long gx = xh_base + 2ull *
            ((((unsigned long long)b * HW + (unsigned)r1) * HW) * IN_PLANES + (unsigned)ic0);
        tdm_load(gx, (unsigned int)(uintptr_t)&XtB[buf][1][XPITCH],
                 IN_PLANES, 256ull, 32, HW, 0ull, 64, 0, 0);
      }
    }
    // out-of-range rows: zero the slab (all threads)
    const int wp = t >> 2, seg = t & 3;
    if (!rin0) *(uint4*)(&XtB[buf][0][(wp + 1) * XPITCH + seg * 8]) = make_uint4(0,0,0,0);
    if (!rin1) *(uint4*)(&XtB[buf][1][(wp + 1) * XPITCH + seg * 8]) = make_uint4(0,0,0,0);
    return 1 + (rin0 ? 1 : 0) + (rin1 ? 1 : 0);
  };

  v8f acc00 = {}, acc01 = {}, acc10 = {}, acc11 = {};
  issue(0, 0);   // prologue: chunk 0 into buffer 0

  const int n = lane & 15, hi = lane >> 4;

  for (int ci = 0; ci < 24; ++ci) {
    const int cur = ci & 1;
    __syncthreads();            // all compute on buf cur^1 finished -> reusable
    int cnt_next = 0;
    if (ci + 1 < 24) cnt_next = issue(ci + 1, cur ^ 1);
    if (wid == 0) {
      // TDMs retire in order; <= cnt_next outstanding ==> chunk ci has landed
      if      (cnt_next == 3) __builtin_amdgcn_s_wait_tensorcnt(3);
      else if (cnt_next == 2) __builtin_amdgcn_s_wait_tensorcnt(2);
      else if (cnt_next == 1) __builtin_amdgcn_s_wait_tensorcnt(1);
      else                    __builtin_amdgcn_s_wait_tensorcnt(0);
    }
    __syncthreads();            // chunk ci visible to all waves

    const _Float16* Afc  = &AfB[cur][0];
    const _Float16* Xtc0 = &XtB[cur][0][0];
    const _Float16* Xtc1 = &XtB[cur][1][0];
#pragma unroll
    for (int kw = 0; kw < 3; ++kw) {
      // A frag (16x32 f16): lane(m=n,hi) reads row k = 16*hi + j contiguous
      // -- reused by all 4 WMMAs.
      const _Float16* arow = Afc + (kw * 64 + mt * 16 + n) * XPITCH;
      Frag16 fa;
      fa.q[0] = *(const uint4*)(arow + hi * 16);
      fa.q[1] = *(const uint4*)(arow + hi * 16 + 8);
      v16h a  = fa.v;
      // B frags (32x16): lane holds col n, k = 16*hi + j contiguous
      v16h b00 = ldsLoadFrag(Xtc0 + (np + n + kw) * XPITCH + hi * 16);
      v16h b01 = ldsLoadFrag(Xtc0 + (np + 16 + n + kw) * XPITCH + hi * 16);
      acc00 = __builtin_amdgcn_wmma_f32_16x16x32_f16(
          false, a, false, b00, (short)0, acc00, false, false);
      acc01 = __builtin_amdgcn_wmma_f32_16x16x32_f16(
          false, a, false, b01, (short)0, acc01, false, false);
      v16h b10 = ldsLoadFrag(Xtc1 + (np + n + kw) * XPITCH + hi * 16);
      v16h b11 = ldsLoadFrag(Xtc1 + (np + 16 + n + kw) * XPITCH + hi * 16);
      acc10 = __builtin_amdgcn_wmma_f32_16x16x32_f16(
          false, a, false, b10, (short)0, acc10, false, false);
      acc11 = __builtin_amdgcn_wmma_f32_16x16x32_f16(
          false, a, false, b11, (short)0, acc11, false, false);
    }
  }

  // store: C/D layout: lane col n = lane&15, row m = v + 8*(lane>>4)
  float* ob0 = out + (((size_t)b * OUT_PLANES + oct * 64 + mt * 16) * HW + h0) * HW;
  float* ob1 = ob0 + HW;  // next output row
#pragma unroll
  for (int v = 0; v < 8; ++v) {
    const int m = v + 8 * hi;
    ob0[(size_t)m * NPIX + np + n]      = acc00[v];
    ob0[(size_t)m * NPIX + np + 16 + n] = acc01[v];
    ob1[(size_t)m * NPIX + np + n]      = acc10[v];
    ob1[(size_t)m * NPIX + np + 16 + n] = acc11[v];
  }
}

// ---------------------------------------------------------------------------
// launch
// ---------------------------------------------------------------------------
extern "C" void kernel_launch(void* const* d_in, const int* in_sizes, int n_in,
                              void* d_out, int out_size, void* d_ws, size_t ws_size,
                              hipStream_t stream) {
  (void)in_sizes; (void)n_in; (void)out_size; (void)ws_size;
  const float* x  = (const float*)d_in[0];
  const float* w  = (const float*)d_in[1];
  const float* w1 = (const float*)d_in[2];
  const float* w2 = (const float*)d_in[3];
  const float* b2 = (const float*)d_in[4];
  float* out = (float*)d_out;

  // workspace: att (1KB) | wq f16 (18.87MB) | xh f16 (33.55MB)
  char* ws = (char*)d_ws;
  float*    att = (float*)ws;
  _Float16* wq  = (_Float16*)(ws + 1024);
  const size_t WQ_BYTES = (size_t)NB * 9 * OUT_PLANES * IN_PLANES * sizeof(_Float16);
  _Float16* xh  = (_Float16*)(ws + 1024 + WQ_BYTES);

  attention_kernel<<<NB, 256, 0, stream>>>(x, w1, w2, b2, att);
  aggregate_kernel<<<dim3(OUT_PLANES, NB), 256, 0, stream>>>(w, att, wq);
  to_nhwc_f16<<<dim3(HW, IN_PLANES / 64, NB), 256, 0, stream>>>(x, xh);
  dynconv_wmma_kernel<<<dim3(HW / 2, OUT_PLANES / 64, NB), 256, 0, stream>>>(xh, wq, out);
}
